// Net_66932770341052
// MI455X (gfx1250) — compile-verified, matrix operand order checked by reading, not run on hardware
//
#include <hip/hip_runtime.h>
#include <hip/hip_bf16.h>
#include <math.h>

// ---------------------------------------------------------------------------
// GAT 2-layer forward for MI455X (gfx1250, wave32).
//   Layer1 GEMM: TDM (tensor_load_to_lds) double-buffered A staging in LDS +
//                split-bf16 WMMA (3x v_wmma_f32_16x16x32_bf16 per 32-K slab),
//                B operands pre-packed to fragment-ordered bf16 hi/lo.
//   Edge phases: segment-softmax via f32 atomics, wave-per-edge aggregation.
// ---------------------------------------------------------------------------

typedef __attribute__((ext_vector_type(16))) __bf16        v16bf;
typedef __attribute__((ext_vector_type(8)))  float         v8f;
typedef __attribute__((ext_vector_type(4)))  unsigned int  u32x4;
typedef __attribute__((ext_vector_type(8)))  unsigned int  u32x8;

#define N_NODES 50000
#define F_IN    256
#define HEADS   2
#define C1      64
#define NCLS    2
#define F1      (HEADS * C1)          // 128
#define XH2W    (HEADS * NCLS)        // 4
#define NSLAB   (F_IN / 32)           // 8 K-slabs
#define NEG_SLOPE 0.2f

__device__ __forceinline__ float lrelu(float x) { return x > 0.f ? x : NEG_SLOPE * x; }

// K index inside a 32-wide K slab for fragment element p (0..15), per the
// CDNA5 ISA 7.12.2 16-bit operand layout: lanes 0-15 cover K {0..7,16..23},
// lanes 16-31 cover K {8..15,24..31}; element 2v/2v+1 = lo/hi half of VGPR v.
__device__ __forceinline__ int frag_k(int p, int lane) {
    int v = p >> 1, h = p & 1;
    int base = ((lane >> 4) & 1) * 8;
    return (v < 4) ? (base + 2 * v + h) : (base + 16 + 2 * (v - 4) + h);
}

// ---------------------------------------------------------------------------
// TDM: async-copy one 16x32 f32 tile (rows of x) into LDS.
// D# group0: count=1, lds_addr, 57-bit global addr, type=2 ("image").
// D# group1: data_size=4B, tensor_dim0=256, tensor_dim1=16, tile=32x16,
//            tensor_dim0_stride=256.  Groups 2/3: zero (dims unused, 2D tile).
// Issued by one wave; completion via s_wait_tensorcnt + workgroup barrier.
// ---------------------------------------------------------------------------
__device__ __forceinline__ void tdm_load_tile(const float* gsrc, unsigned lds_off) {
    unsigned long long ga = (unsigned long long)(uintptr_t)gsrc;
    u32x4 g0 = { 1u,                                   // count=1, user mode
                 lds_off,                              // lds_addr (bytes)
                 (unsigned)ga,                         // global_addr[31:0]
                 (unsigned)((ga >> 32) & 0x01FFFFFFu)  // global_addr[56:32]
                     | 0x80000000u };                  // type=2 at [127:126]
    u32x8 g1 = { 0x00020000u,                          // data_size=2 (4 bytes)
                 (256u & 0xFFFFu) << 16,               // tensor_dim0 lo16 = 256
                 (16u << 16),                          // dim0 hi | tensor_dim1 = 16
                 (32u << 16),                          // dim1 hi | tile_dim0 = 32
                 16u,                                  // tile_dim1=16, tile_dim2=0
                 256u,                                 // tensor_dim0_stride lo32
                 0u, 0u };
    u32x4 gz = { 0u, 0u, 0u, 0u };
    asm volatile("tensor_load_to_lds %0, %1, %2, %3"
                 :: "s"(g0), "s"(g1), "s"(gz), "s"(gz) : "memory");
}

// ---------------------------------------------------------------------------
// Kernel 0: initialize accumulators in workspace.
// ---------------------------------------------------------------------------
__global__ void k_init(float* out1, float* out2, float* den1, float* den2,
                       float* m1, float* m2) {
    long t = (long)blockIdx.x * blockDim.x + threadIdx.x;
    if (t < (long)N_NODES * F1) out1[t] = 0.f;
    if (t < (long)N_NODES * XH2W) out2[t] = 0.f;
    if (t < (long)N_NODES * HEADS) {
        den1[t] = 0.f; den2[t] = 0.f;
        m1[t] = -INFINITY; m2[t] = -INFINITY;
    }
}

// ---------------------------------------------------------------------------
// Kernel: pre-pack W1 into fragment-ordered bf16 hi/lo (one (slab,jtile,lane)
// per thread -> 16 contiguous bf16 = one 32 B v16bf load in the GEMM). 128 KB total.
// ---------------------------------------------------------------------------
__global__ void k_packW(const float* __restrict__ W,
                        __bf16* __restrict__ Whi, __bf16* __restrict__ Wlo) {
    int t = blockIdx.x * blockDim.x + threadIdx.x;   // (slab*8 + j)*32 + lane
    if (t >= NSLAB * 8 * 32) return;
    int lane = t & 31;
    int j    = (t >> 5) & 7;
    int slab = t >> 8;
    int col  = j * 16 + (lane & 15);
#pragma unroll
    for (int p = 0; p < 16; ++p) {
        int k = slab * 32 + frag_k(p, lane);
        float w = W[k * F1 + col];
        __bf16 hi = (__bf16)w;
        Whi[(size_t)t * 16 + p] = hi;
        Wlo[(size_t)t * 16 + p] = (__bf16)(w - (float)hi);
    }
}

// ---------------------------------------------------------------------------
// Kernel 1: xh1 = x @ W1.  Block = 256 threads = 8 waves; block owns 16 rows,
// wave w owns column tile w.  A tile staged by TDM into double-buffered LDS.
// ---------------------------------------------------------------------------
__global__ void __launch_bounds__(256)
k_gemm1_wmma(const float* __restrict__ x,
             const __bf16* __restrict__ Whi, const __bf16* __restrict__ Wlo,
             float* __restrict__ xh1) {
    __shared__ __align__(128) float tile[2][16 * 32];
    const int tid  = threadIdx.x;
    const int lane = tid & 31;
    const int jt   = tid >> 5;                  // column tile 0..7
    const int i0   = blockIdx.x * 16;           // row tile
    const int m    = lane & 15;

    // prologue: wave 0 kicks off the first TDM tile copy
    if (tid < 32)
        tdm_load_tile(x + (size_t)i0 * F_IN,
                      (unsigned)(uintptr_t)&tile[0][0]);

    v8f acc = {};
    for (int slab = 0; slab < NSLAB; ++slab) {
        __builtin_amdgcn_s_wait_tensorcnt(0);   // no-op for non-issuing waves
        __syncthreads();                        // staged tile visible to all
        if (slab + 1 < NSLAB && tid < 32)       // prefetch next slab (overlaps)
            tdm_load_tile(x + (size_t)i0 * F_IN + (slab + 1) * 32,
                          (unsigned)(uintptr_t)&tile[(slab + 1) & 1][0]);

        const float* arow = &tile[slab & 1][0];
        v16bf ah, al;
#pragma unroll
        for (int p = 0; p < 16; ++p) {          // split f32 -> hi + residual
            float a   = arow[m * 32 + frag_k(p, lane)];
            __bf16 hi = (__bf16)a;
            ah[p] = hi;
            al[p] = (__bf16)(a - (float)hi);
        }
        const size_t fb = ((size_t)(slab * 8 + jt) * 32 + lane) * 16;
        v16bf bh = *(const v16bf*)(Whi + fb);
        v16bf bl = *(const v16bf*)(Wlo + fb);
        // hi*hi + lo*hi + hi*lo  ~= full f32 product
        acc = __builtin_amdgcn_wmma_f32_16x16x32_bf16(false, ah, false, bh,
                                                      (short)0, acc, false, false);
        acc = __builtin_amdgcn_wmma_f32_16x16x32_bf16(false, al, false, bh,
                                                      (short)0, acc, false, false);
        acc = __builtin_amdgcn_wmma_f32_16x16x32_bf16(false, ah, false, bl,
                                                      (short)0, acc, false, false);
        __syncthreads();                        // reads done before overwrite
    }
    // D layout: VGPR r, lanes 0-15 -> M=r, lanes 16-31 -> M=8+r; N=lane&15.
    const int n    = lane & 15;
    const int mrow = (lane >> 4) * 8;
#pragma unroll
    for (int r = 0; r < 8; ++r)
        xh1[(i0 + mrow + r) * F1 + (jt * 16) + n] = acc[r];
}

// ---------------------------------------------------------------------------
// Kernel 2: per-node attention logits, layer 1. Thread per (node, head).
// ---------------------------------------------------------------------------
__global__ void k_att1(const float* __restrict__ xh1,
                       const float* __restrict__ att_src,
                       const float* __restrict__ att_dst,
                       float* __restrict__ asrc, float* __restrict__ adst) {
    int t = blockIdx.x * blockDim.x + threadIdx.x;
    if (t >= N_NODES * HEADS) return;
    int n = t >> 1, h = t & 1;
    const float* row = xh1 + (long)n * F1 + h * C1;
    const float* as  = att_src + h * C1;
    const float* ad  = att_dst + h * C1;
    float s = 0.f, d = 0.f;
#pragma unroll 8
    for (int c = 0; c < C1; ++c) { float v = row[c]; s += v * as[c]; d += v * ad[c]; }
    asrc[t] = s; adst[t] = d;
}

// ---------------------------------------------------------------------------
// Edge pass 1: segment max (thread per edge; self-loops are e >= E).
// ---------------------------------------------------------------------------
__global__ void k_edge_max(const int* __restrict__ src_e, const int* __restrict__ dst_e,
                           int E, int Etot,
                           const float* __restrict__ asrc, const float* __restrict__ adst,
                           float* __restrict__ m) {
    int e = blockIdx.x * blockDim.x + threadIdx.x;
    if (e >= Etot) return;
    int s = (e < E) ? src_e[e] : (e - E);
    int d = (e < E) ? dst_e[e] : (e - E);
#pragma unroll
    for (int h = 0; h < HEADS; ++h) {
        float ev = lrelu(asrc[s * HEADS + h] + adst[d * HEADS + h]);
        atomicMax(&m[d * HEADS + h], ev);
    }
}

// Edge pass 2: segment sum of exp(e - max).
__global__ void k_edge_sum(const int* __restrict__ src_e, const int* __restrict__ dst_e,
                           int E, int Etot,
                           const float* __restrict__ asrc, const float* __restrict__ adst,
                           const float* __restrict__ m, float* __restrict__ den) {
    int e = blockIdx.x * blockDim.x + threadIdx.x;
    if (e >= Etot) return;
    int s = (e < E) ? src_e[e] : (e - E);
    int d = (e < E) ? dst_e[e] : (e - E);
#pragma unroll
    for (int h = 0; h < HEADS; ++h) {
        float ev = lrelu(asrc[s * HEADS + h] + adst[d * HEADS + h]);
        atomicAdd(&den[d * HEADS + h], expf(ev - m[d * HEADS + h]));
    }
}

// ---------------------------------------------------------------------------
// Edge pass 3, layer 1: alpha-weighted scatter of xh1[src] into out1[dst].
// One wave32 per edge; per-head alpha computed ONCE per lane (2 exps), then
// 32 lanes sweep the 128-channel row in 4 coalesced 128 B steps.
// ---------------------------------------------------------------------------
__global__ void k_edge_agg1(const int* __restrict__ src_e, const int* __restrict__ dst_e,
                            int E, int Etot,
                            const float* __restrict__ asrc, const float* __restrict__ adst,
                            const float* __restrict__ m, const float* __restrict__ den,
                            const float* __restrict__ xh1, float* __restrict__ out1) {
    int wave = (blockIdx.x * blockDim.x + threadIdx.x) >> 5;
    int lane = threadIdx.x & 31;
    if (wave >= Etot) return;
    int s = (wave < E) ? src_e[wave] : (wave - E);
    int d = (wave < E) ? dst_e[wave] : (wave - E);
    float ev0 = lrelu(asrc[s * HEADS + 0] + adst[d * HEADS + 0]);
    float a0  = expf(ev0 - m[d * HEADS + 0]) / den[d * HEADS + 0];
    float ev1 = lrelu(asrc[s * HEADS + 1] + adst[d * HEADS + 1]);
    float a1  = expf(ev1 - m[d * HEADS + 1]) / den[d * HEADS + 1];
#pragma unroll
    for (int it = 0; it < 4; ++it) {
        int   f     = it * 32 + lane;        // channel 0..127 (head = f>>6)
        float alpha = (it < 2) ? a0 : a1;
        atomicAdd(&out1[(long)d * F1 + f], alpha * xh1[(long)s * F1 + f]);
    }
}

// Kernel: embeddings = elu(out1 + b1)  (written straight into d_out region 2)
__global__ void k_final1(const float* __restrict__ out1, const float* __restrict__ b1,
                         float* __restrict__ emb) {
    long t = (long)blockIdx.x * blockDim.x + threadIdx.x;
    if (t >= (long)N_NODES * F1) return;
    float v = out1[t] + b1[t & (F1 - 1)];
    emb[t] = v > 0.f ? v : expm1f(v);
}

// ---------------------------------------------------------------------------
// Kernel: xh2 = emb @ W2 (128 -> 4) fused with layer-2 attention logits.
// ---------------------------------------------------------------------------
__global__ void k_gemm2_att2(const float* __restrict__ emb, const float* __restrict__ W2,
                             const float* __restrict__ att_src, const float* __restrict__ att_dst,
                             float* __restrict__ xh2,
                             float* __restrict__ asrc, float* __restrict__ adst) {
    int n = blockIdx.x * blockDim.x + threadIdx.x;
    if (n >= N_NODES) return;
    float v[XH2W];
#pragma unroll
    for (int j = 0; j < XH2W; ++j) v[j] = 0.f;
    const float* row = emb + (long)n * F1;
    for (int k = 0; k < F1; ++k) {
        float r = row[k];
#pragma unroll
        for (int j = 0; j < XH2W; ++j) v[j] += r * W2[k * XH2W + j];
    }
#pragma unroll
    for (int j = 0; j < XH2W; ++j) xh2[(long)n * XH2W + j] = v[j];
#pragma unroll
    for (int h = 0; h < HEADS; ++h) {
        float s = 0.f, d = 0.f;
#pragma unroll
        for (int c = 0; c < NCLS; ++c) {
            s += v[h * NCLS + c] * att_src[h * NCLS + c];
            d += v[h * NCLS + c] * att_dst[h * NCLS + c];
        }
        asrc[n * HEADS + h] = s; adst[n * HEADS + h] = d;
    }
}

// Edge pass 3, layer 2 (C=2): thread per edge, 4 scattered adds.
__global__ void k_edge_agg2(const int* __restrict__ src_e, const int* __restrict__ dst_e,
                            int E, int Etot,
                            const float* __restrict__ asrc, const float* __restrict__ adst,
                            const float* __restrict__ m, const float* __restrict__ den,
                            const float* __restrict__ xh2, float* __restrict__ out2) {
    int e = blockIdx.x * blockDim.x + threadIdx.x;
    if (e >= Etot) return;
    int s = (e < E) ? src_e[e] : (e - E);
    int d = (e < E) ? dst_e[e] : (e - E);
#pragma unroll
    for (int h = 0; h < HEADS; ++h) {
        float ev    = lrelu(asrc[s * HEADS + h] + adst[d * HEADS + h]);
        float alpha = expf(ev - m[d * HEADS + h]) / den[d * HEADS + h];
#pragma unroll
        for (int c = 0; c < NCLS; ++c)
            atomicAdd(&out2[(long)d * XH2W + h * NCLS + c],
                      alpha * xh2[(long)s * XH2W + h * NCLS + c]);
    }
}

// Kernel: out = sigmoid(mean_heads(out2) + b2)
__global__ void k_final2(const float* __restrict__ out2, const float* __restrict__ b2,
                         float* __restrict__ out) {
    int t = blockIdx.x * blockDim.x + threadIdx.x;
    if (t >= N_NODES * NCLS) return;
    int n = t / NCLS, c = t % NCLS;
    float v = 0.5f * (out2[(long)n * XH2W + c] + out2[(long)n * XH2W + NCLS + c]) + b2[c];
    out[t] = 1.f / (1.f + expf(-v));
}

// ---------------------------------------------------------------------------
extern "C" void kernel_launch(void* const* d_in, const int* in_sizes, int n_in,
                              void* d_out, int out_size, void* d_ws, size_t ws_size,
                              hipStream_t stream) {
    const float* x        = (const float*)d_in[0];
    const int*   eidx     = (const int*)  d_in[1];
    const float* W1       = (const float*)d_in[2];
    const float* att_src1 = (const float*)d_in[3];
    const float* att_dst1 = (const float*)d_in[4];
    const float* b1       = (const float*)d_in[5];
    const float* W2       = (const float*)d_in[6];
    const float* att_src2 = (const float*)d_in[7];
    const float* att_dst2 = (const float*)d_in[8];
    const float* b2       = (const float*)d_in[9];

    const int E    = in_sizes[1] / 2;      // 1.6M
    const int Etot = E + N_NODES;          // + self loops
    const int* src_e = eidx;
    const int* dst_e = eidx + E;

    // d_out = [out (N*2) | embeddings (N*128)]
    float* out = (float*)d_out;
    float* emb = out + (long)N_NODES * NCLS;

    // workspace carve-up (~56 MB of f32 + 128 KB packed bf16 W1)
    float* p     = (float*)d_ws;
    float* xh1   = p;  p += (long)N_NODES * F1;
    float* out1  = p;  p += (long)N_NODES * F1;
    float* asrc1 = p;  p += N_NODES * HEADS;
    float* adst1 = p;  p += N_NODES * HEADS;
    float* m1    = p;  p += N_NODES * HEADS;
    float* den1  = p;  p += N_NODES * HEADS;
    float* xh2   = p;  p += N_NODES * XH2W;
    float* asrc2 = p;  p += N_NODES * HEADS;
    float* adst2 = p;  p += N_NODES * HEADS;
    float* m2    = p;  p += N_NODES * HEADS;
    float* den2  = p;  p += N_NODES * HEADS;
    float* out2  = p;  p += N_NODES * XH2W;
    __bf16* Whi  = (__bf16*)p;                     // 32 KB elements (64 KB)
    __bf16* Wlo  = Whi + (size_t)NSLAB * 8 * 32 * 16;
    (void)ws_size; (void)n_in; (void)out_size;

    const int B = 256;
    // init accumulators + pack W1 fragments
    k_init<<<((long)N_NODES * F1 + B - 1) / B, B, 0, stream>>>(out1, out2, den1, den2, m1, m2);
    k_packW<<<(NSLAB * 8 * 32 + B - 1) / B, B, 0, stream>>>(W1, Whi, Wlo);
    // layer 1 dense
    k_gemm1_wmma<<<N_NODES / 16, B, 0, stream>>>(x, Whi, Wlo, xh1);
    k_att1<<<(N_NODES * HEADS + B - 1) / B, B, 0, stream>>>(xh1, att_src1, att_dst1, asrc1, adst1);
    // layer 1 segment softmax + aggregate
    k_edge_max<<<(Etot + B - 1) / B, B, 0, stream>>>(src_e, dst_e, E, Etot, asrc1, adst1, m1);
    k_edge_sum<<<(Etot + B - 1) / B, B, 0, stream>>>(src_e, dst_e, E, Etot, asrc1, adst1, m1, den1);
    k_edge_agg1<<<((long)Etot * 32 + B - 1) / B, B, 0, stream>>>(src_e, dst_e, E, Etot,
                                                                 asrc1, adst1, m1, den1, xh1, out1);
    k_final1<<<((long)N_NODES * F1 + B - 1) / B, B, 0, stream>>>(out1, b1, emb);
    // layer 2 dense + logits
    k_gemm2_att2<<<(N_NODES + B - 1) / B, B, 0, stream>>>(emb, W2, att_src2, att_dst2,
                                                          xh2, asrc2, adst2);
    // layer 2 segment softmax + aggregate
    k_edge_max<<<(Etot + B - 1) / B, B, 0, stream>>>(src_e, dst_e, E, Etot, asrc2, adst2, m2);
    k_edge_sum<<<(Etot + B - 1) / B, B, 0, stream>>>(src_e, dst_e, E, Etot, asrc2, adst2, m2, den2);
    k_edge_agg2<<<(Etot + B - 1) / B, B, 0, stream>>>(src_e, dst_e, E, Etot,
                                                      asrc2, adst2, m2, den2, xh2, out2);
    k_final2<<<(N_NODES * NCLS + B - 1) / B, B, 0, stream>>>(out2, b2, out);
}
